// OpenBT5Attention_87196426043677
// MI455X (gfx1250) — matrix operand order repaired and checked by reading, not compiled, not run on hardware
//
#include <hip/hip_runtime.h>
#include <hip/hip_bf16.h>

#define H_   32
#define D_   64
#define HID_ 2048
#define S_   2048
#define B_   2

typedef __attribute__((ext_vector_type(16))) __bf16 v16bf;
typedef __attribute__((ext_vector_type(4)))  __bf16 v4bf;
typedef __attribute__((ext_vector_type(8)))  float  v8f;

// ---------------------------------------------------------------------------
// TDM support (gfx1250 Tensor Data Mover) — guarded so the host pass and
// builds without the builtin fall back to cooperative vector loads.
// ---------------------------------------------------------------------------
#if defined(__HIP_DEVICE_COMPILE__) && __has_builtin(__builtin_amdgcn_tensor_load_to_lds)
#define TDM_OK 1
#else
#define TDM_OK 0
#endif

#if TDM_OK
typedef unsigned int u32x4 __attribute__((ext_vector_type(4)));
typedef int          i32x4 __attribute__((ext_vector_type(4)));
typedef int          i32x8 __attribute__((ext_vector_type(8)));

__device__ __forceinline__ unsigned lds_addr_of(const void* p) {
  typedef __attribute__((address_space(3))) const void* lds_ptr_t;
  return (unsigned)(unsigned long long)(lds_ptr_t)p;
}

// 2-D tile load via TDM.  dsize: 0=1B,1=2B,2=4B.  Rows of tile0 elements,
// tile1 rows; tensor row stride stride0 (elements).  LDS destination rows are
// padded: pad_interval code pi (DWORDs = 2<<pi), pad_amount code pa (pa+1 DW).
__device__ __forceinline__ void tdm_load_2d(unsigned lds, unsigned long long ga,
                                            int tile0, int tile1,
                                            long long tdim0, long long tdim1,
                                            long long stride0,
                                            int dsize, int pi, int pa) {
  u32x4 g0;
  g0[0] = 1u;                                            // count=1, user mode
  g0[1] = lds;                                           // lds_addr
  g0[2] = (unsigned)ga;                                  // global_addr[31:0]
  g0[3] = (unsigned)((ga >> 32) & 0x1FFFFFFu) | (2u << 30); // addr hi | type=2
  i32x8 g1;
  g1[0] = (dsize << 16) | (1 << 20) | (pi << 22) | (pa << 25);
  g1[1] = (int)((tdim0 & 0xFFFFll) << 16);               // tensor_dim0 lo
  g1[2] = (int)(((tdim0 >> 16) & 0xFFFF) | ((tdim1 & 0xFFFF) << 16));
  g1[3] = (int)(((tdim1 >> 16) & 0xFFFF) | ((long long)(tile0 & 0xFFFF) << 16));
  g1[4] = (int)(tile1 & 0xFFFF);                         // tile_dim1 (tile_dim2=0)
  g1[5] = (int)(stride0 & 0xFFFFFFFFll);                 // tensor_dim0_stride lo
  g1[6] = (int)((stride0 >> 32) & 0xFFFF);               // stride hi (dim1_stride=0)
  g1[7] = 0;
  i32x4 z4 = {0, 0, 0, 0};
#if __clang_major__ >= 23
  i32x8 z8 = {0, 0, 0, 0, 0, 0, 0, 0};
  __builtin_amdgcn_tensor_load_to_lds(g0, g1, z4, z4, z8, 0);
#else
  __builtin_amdgcn_tensor_load_to_lds(g0, g1, z4, z4, 0);
#endif
}
#endif // TDM_OK

// ---- WMMA fragment loaders (CDNA5 wave32 layouts, CDNA5 ISA §7.12.2) ----
__device__ __forceinline__ v16bf load_a_frag(const __bf16* p, int ld) {
  const int lane = threadIdx.x & 31;
  const int r  = lane & 15;
  const int kb = (lane >> 4) << 3;
  v16bf a;
#pragma unroll
  for (int j = 0; j < 8; ++j) {
    const int k = ((j & 4) << 2) + kb + ((j & 3) << 1);
    a[2*j]     = p[r*ld + k];
    a[2*j + 1] = p[r*ld + k + 1];
  }
  return a;
}

__device__ __forceinline__ v16bf load_b_frag(const __bf16* p, int ld) {
  const int lane = threadIdx.x & 31;
  const int n  = lane & 15;
  const int kb = (lane >> 4) << 3;
  v16bf b;
#pragma unroll
  for (int j = 0; j < 8; ++j) {
    const int k = ((j & 4) << 2) + kb + ((j & 3) << 1);
    b[2*j]     = p[n*ld + k];
    b[2*j + 1] = p[n*ld + k + 1];
  }
  return b;
}

__device__ __forceinline__ v8f wmma_bf16(v16bf a, v16bf b, v8f c) {
  return __builtin_amdgcn_wmma_f32_16x16x32_bf16(false, a, false, b, (short)0, c,
                                                 false, false);
}

// ============================================================================
// Kernel 1: qkv = hidden @ Wqkv^T, RoPE(q,k), fold 1/sqrt(D) into q.
// (fp32 sources need conversion, so tiles are staged with vector loads.)
// ============================================================================
__global__ __launch_bounds__(256)
void qkv_rope_kernel(const float* __restrict__ hidden,
                     const int*   __restrict__ pos_ids,
                     const float* __restrict__ Wqkv,
                     __bf16* __restrict__ Qr,   // [B,H,S,D]
                     __bf16* __restrict__ Kr,   // [B,H,S,D]
                     __bf16* __restrict__ Vt) { // [B,H,D,S]
  __shared__ __align__(16) __bf16 ldsA[128][40];
  __shared__ __align__(16) __bf16 ldsB[64][40];
  __shared__ __align__(16) float  ldsO[128][68];

  const int tid  = threadIdx.x;
  const int lane = tid & 31;
  const int wave = tid >> 5;
  const int wm = wave & 3, wn = wave >> 2;
  const int m0 = blockIdx.y * 128;
  const int n0 = blockIdx.x * 64;
  const int hi = lane >> 4, nc = lane & 15;

  v8f acc[2][2] = {};

  for (int kt = 0; kt < HID_; kt += 32) {
    if (kt + 32 < HID_) {
      __builtin_prefetch(hidden + (size_t)(m0 + (tid >> 1))*HID_ + kt + 32 + (tid & 1)*16, 0, 1);
      if (tid < 128)
        __builtin_prefetch(Wqkv + (size_t)(n0 + (tid >> 1))*HID_ + kt + 32 + (tid & 1)*16, 0, 1);
    }
#pragma unroll
    for (int i = 0; i < 4; ++i) {            // A: 128x32 f32 as float4
      int idx = i*256 + tid, r = idx >> 3, c4 = idx & 7;
      float4 v = ((const float4*)(hidden + (size_t)(m0 + r)*HID_ + kt))[c4];
      v4bf w; w[0] = (__bf16)v.x; w[1] = (__bf16)v.y; w[2] = (__bf16)v.z; w[3] = (__bf16)v.w;
      *(v4bf*)&ldsA[r][c4*4] = w;
    }
#pragma unroll
    for (int i = 0; i < 2; ++i) {            // B: 64x32 f32 as float4
      int idx = i*256 + tid, r = idx >> 3, c4 = idx & 7;
      float4 v = ((const float4*)(Wqkv + (size_t)(n0 + r)*HID_ + kt))[c4];
      v4bf w; w[0] = (__bf16)v.x; w[1] = (__bf16)v.y; w[2] = (__bf16)v.z; w[3] = (__bf16)v.w;
      *(v4bf*)&ldsB[r][c4*4] = w;
    }
    __syncthreads();
    v16bf a0 = load_a_frag(&ldsA[wm*32     ][0], 40);
    v16bf a1 = load_a_frag(&ldsA[wm*32 + 16][0], 40);
    v16bf b0 = load_b_frag(&ldsB[wn*32     ][0], 40);
    v16bf b1 = load_b_frag(&ldsB[wn*32 + 16][0], 40);
    acc[0][0] = wmma_bf16(a0, b0, acc[0][0]);
    acc[0][1] = wmma_bf16(a0, b1, acc[0][1]);
    acc[1][0] = wmma_bf16(a1, b0, acc[1][0]);
    acc[1][1] = wmma_bf16(a1, b1, acc[1][1]);
    __syncthreads();
  }

#pragma unroll
  for (int mi = 0; mi < 2; ++mi)
#pragma unroll
    for (int ni = 0; ni < 2; ++ni)
#pragma unroll
      for (int r = 0; r < 8; ++r)
        ldsO[wm*32 + mi*16 + r + 8*hi][wn*32 + ni*16 + nc] = acc[mi][ni][r];
  __syncthreads();

  const int hh  = n0 / 192;
  const int seg = (n0 % 192) >> 6;           // 0=q 1=k 2=v
#pragma unroll
  for (int i = 0; i < 32; ++i) {
    int e = i*256 + tid, r = e >> 6, d = e & 63;
    int m = m0 + r, b = m >> 11, s = m & (S_ - 1);
    float x = ldsO[r][d];
    if (seg == 2) {
      Vt[(((size_t)b*H_ + hh)*D_ + d)*S_ + s] = (__bf16)x;
    } else {
      int   p  = pos_ids[m];
      int   i2 = d & 31;
      float freq = __expf(-0.28782313662425572f * (float)i2); // 10000^(-2i/64)
      float t    = (float)p * freq;
      float sn   = __sinf(t), cs = __cosf(t);
      float xp   = ldsO[r][d ^ 32];
      float v    = (d < 32) ? (x*cs - xp*sn) : (x*cs + xp*sn);
      if (seg == 0) v *= 0.125f;
      __bf16* dst = (seg == 0) ? Qr : Kr;
      dst[(((size_t)b*H_ + hh)*S_ + s)*D_ + d] = (__bf16)v;
    }
  }
}

// ============================================================================
// Kernel 2: flash-style attention; K/V/mask tiles staged by the TDM.
// ============================================================================
__global__ __launch_bounds__(256)
void attention_kernel(const __bf16* __restrict__ Qr,
                      const __bf16* __restrict__ Kr,
                      const __bf16* __restrict__ Vt,
                      const unsigned char* __restrict__ mask, // [B,1,S,S]
                      __bf16* __restrict__ ctx) {             // [B,S,H*D]
  __shared__ __align__(16) __bf16 ldsK[64][72];          // [key][d]
  __shared__ __align__(16) __bf16 ldsV[64][72];          // [d][key]
  __shared__ __align__(16) unsigned char ldsM[64][80];   // [q][key] mask bytes
  __shared__ __align__(16) float  ldsS[64][66];
  __shared__ __align__(16) __bf16 ldsP[64][72];
  __shared__ float row_m[64], row_l[64], row_scale[64];
  __shared__ float pred[64][4];

  const int tid  = threadIdx.x;
  const int lane = tid & 31;
  const int wave = tid >> 5;
  const int wq = wave & 3, wk = wave >> 2;
  const int bh = blockIdx.y;
  const int b  = bh >> 5;
  const int h  = bh & 31;
  const int q0 = blockIdx.x * 64;
  const int hi = lane >> 4, nc = lane & 15;

  const size_t headSD = ((size_t)b*H_ + h) * S_ * D_;
  const size_t headDS = ((size_t)b*H_ + h) * D_ * S_;
  const size_t maskQ0 = ((size_t)b*S_ + q0) * S_;

  if (tid < 64) { row_m[tid] = -1e30f; row_l[tid] = 0.0f; }

  const __bf16* qbase = Qr + headSD + (size_t)(q0 + wq*16) * D_;
  v16bf aq0 = load_a_frag(qbase,      64);
  v16bf aq1 = load_a_frag(qbase + 32, 64);

  v8f oacc[2] = {};

  const int srow = tid >> 2;
  const int sq   = tid & 3;

  for (int kt = 0; kt < S_; kt += 64) {
    __syncthreads();
#if TDM_OK
    if (wave == 0) {
      // K tile: rows kt..kt+63 of [S,D]    (2B elems, 32+4 DW rows)
      tdm_load_2d(lds_addr_of(&ldsK[0][0]),
                  (unsigned long long)(Kr + headSD + (size_t)kt * D_),
                  64, 64, D_, S_, D_, 1, 4, 3);
      // V tile: 64 d-rows of [D,S], cols kt..kt+63
      tdm_load_2d(lds_addr_of(&ldsV[0][0]),
                  (unsigned long long)(Vt + headDS + kt),
                  64, 64, S_, D_, S_, 1, 4, 3);
      // mask tile: 64 q-rows x 64 key-bytes (1B elems, 16+4 DW rows)
      tdm_load_2d(lds_addr_of(&ldsM[0][0]),
                  (unsigned long long)(mask + maskQ0 + kt),
                  64, 64, S_, S_, S_, 0, 3, 3);
      __builtin_amdgcn_s_wait_tensorcnt(0);
    }
#else
#pragma unroll
    for (int i = 0; i < 16; ++i) {
      int e = i*256 + tid, r = e >> 6, c = e & 63;
      ldsK[r][c] = Kr[headSD + (size_t)(kt + r)*D_ + c];
      ldsV[r][c] = Vt[headDS + (size_t)r*S_ + kt + c];
      ldsM[r][c] = mask[maskQ0 + (size_t)r*S_ + kt + c];
    }
#endif
    if (kt + 64 < S_) {
      __builtin_prefetch(Kr + headSD + (size_t)(kt + 64 + (tid >> 2))*D_, 0, 1);
      __builtin_prefetch(Vt + headDS + (size_t)(tid >> 2)*S_ + kt + 64, 0, 1);
    }
    __syncthreads();

    // ---- scores: wave does 16q x 32k, K-dim = D = 64 ----
#pragma unroll
    for (int nf = 0; nf < 2; ++nf) {
      int kcol = wk*32 + nf*16;
      v8f sc = {};
      v16bf bk0 = load_b_frag(&ldsK[kcol][0],  72);
      v16bf bk1 = load_b_frag(&ldsK[kcol][32], 72);
      sc = wmma_bf16(aq0, bk0, sc);
      sc = wmma_bf16(aq1, bk1, sc);
#pragma unroll
      for (int r = 0; r < 8; ++r) {
        int row = wq*16 + r + 8*hi;
        int col = kcol + nc;
        ldsS[row][col] = ldsM[row][col] ? -10000.0f : sc[r];
      }
    }
    __syncthreads();

    // ---- online softmax ----
    {
      float lm = -1e30f;
#pragma unroll
      for (int c = 0; c < 16; ++c) lm = fmaxf(lm, ldsS[srow][sq*16 + c]);
      pred[srow][sq] = lm;
    }
    __syncthreads();
    if (sq == 0) {
      float mo = row_m[srow];
      float mn = fmaxf(fmaxf(pred[srow][0], pred[srow][1]),
                       fmaxf(pred[srow][2], pred[srow][3]));
      mn = fmaxf(mo, mn);
      row_scale[srow] = __expf(mo - mn);
      row_m[srow] = mn;
    }
    __syncthreads();
    {
      float mn = row_m[srow];
      float sum = 0.0f;
#pragma unroll
      for (int c = 0; c < 16; ++c) {
        float e = __expf(ldsS[srow][sq*16 + c] - mn);
        sum += e;
        ldsP[srow][sq*16 + c] = (__bf16)e;
      }
      pred[srow][sq] = sum;
    }
    __syncthreads();
    if (sq == 0) {
      row_l[srow] = row_l[srow]*row_scale[srow]
                  + pred[srow][0] + pred[srow][1] + pred[srow][2] + pred[srow][3];
    }
    __syncthreads();

    // ---- rescale O and accumulate P @ V ----
    v16bf ap0 = load_a_frag(&ldsP[wq*16][0],  72);
    v16bf ap1 = load_a_frag(&ldsP[wq*16][32], 72);
#pragma unroll
    for (int ni = 0; ni < 2; ++ni) {
      int dcol = wk*32 + ni*16;
      v16bf bv0 = load_b_frag(&ldsV[dcol][0],  72);
      v16bf bv1 = load_b_frag(&ldsV[dcol][32], 72);
#pragma unroll
      for (int r = 0; r < 8; ++r)
        oacc[ni][r] = oacc[ni][r] * row_scale[wq*16 + r + 8*hi];
      oacc[ni] = wmma_bf16(ap0, bv0, oacc[ni]);
      oacc[ni] = wmma_bf16(ap1, bv1, oacc[ni]);
    }
  }
  __syncthreads();

#pragma unroll
  for (int ni = 0; ni < 2; ++ni)
#pragma unroll
    for (int r = 0; r < 8; ++r) {
      int row = wq*16 + r + 8*hi;
      int d   = wk*32 + ni*16 + nc;
      float inv = 1.0f / row_l[row];
      ctx[((size_t)b*S_ + (q0 + row))*(H_*D_) + h*D_ + d] =
          (__bf16)(oacc[ni][r] * inv);
    }
}

// ============================================================================
// Kernel 3: out = ctx @ Wo^T  (A tile is bf16 in memory -> staged by TDM)
// ============================================================================
__global__ __launch_bounds__(256)
void out_proj_kernel(const __bf16* __restrict__ ctx,
                     const float*  __restrict__ Wo,
                     float* __restrict__ out) {
  __shared__ __align__(16) __bf16 ldsA[128][40];
  __shared__ __align__(16) __bf16 ldsB[64][40];

  const int tid  = threadIdx.x;
  const int lane = tid & 31;
  const int wave = tid >> 5;
  const int wm = wave & 3, wn = wave >> 2;
  const int m0 = blockIdx.y * 128;
  const int n0 = blockIdx.x * 64;
  const int hi = lane >> 4, nc = lane & 15;

  v8f acc[2][2] = {};

  for (int kt = 0; kt < HID_; kt += 32) {
    if (kt + 32 < HID_ && tid < 128)
      __builtin_prefetch(Wo + (size_t)(n0 + (tid >> 1))*HID_ + kt + 32 + (tid & 1)*16, 0, 1);
#if TDM_OK
    if (wave == 0) {
      // A tile: 128 rows x 32 bf16 (64 B = 16 DW rows, +4 DW pad -> stride 40 bf16)
      tdm_load_2d(lds_addr_of(&ldsA[0][0]),
                  (unsigned long long)(ctx + (size_t)m0*HID_ + kt),
                  32, 128, HID_, (long long)B_*S_, HID_, 1, 3, 3);
    }
#else
#pragma unroll
    for (int i = 0; i < 2; ++i) {            // A: 128x32 bf16 as 128-bit copies
      int idx = i*256 + tid, r = idx >> 2, c8 = idx & 3;
      *(uint4*)&ldsA[r][c8*8] =
          ((const uint4*)(ctx + (size_t)(m0 + r)*HID_ + kt))[c8];
    }
#endif
#pragma unroll
    for (int i = 0; i < 2; ++i) {            // B: 64x32 f32 as float4 + convert
      int idx = i*256 + tid, r = idx >> 3, c4 = idx & 7;
      float4 v = ((const float4*)(Wo + (size_t)(n0 + r)*HID_ + kt))[c4];
      v4bf w; w[0] = (__bf16)v.x; w[1] = (__bf16)v.y; w[2] = (__bf16)v.z; w[3] = (__bf16)v.w;
      *(v4bf*)&ldsB[r][c4*4] = w;
    }
#if TDM_OK
    if (wave == 0) __builtin_amdgcn_s_wait_tensorcnt(0);
#endif
    __syncthreads();
    v16bf a0 = load_a_frag(&ldsA[wm*32     ][0], 40);
    v16bf a1 = load_a_frag(&ldsA[wm*32 + 16][0], 40);
    v16bf b0 = load_b_frag(&ldsB[wn*32     ][0], 40);
    v16bf b1 = load_b_frag(&ldsB[wn*32 + 16][0], 40);
    acc[0][0] = wmma_bf16(a0, b0, acc[0][0]);
    acc[0][1] = wmma_bf16(a0, b1, acc[0][1]);
    acc[1][0] = wmma_bf16(a1, b0, acc[1][0]);
    acc[1][1] = wmma_bf16(a1, b1, acc[1][1]);
    __syncthreads();
  }

#pragma unroll
  for (int mi = 0; mi < 2; ++mi)
#pragma unroll
    for (int ni = 0; ni < 2; ++ni)
#pragma unroll
      for (int r = 0; r < 8; ++r)
        out[(size_t)(m0 + wm*32 + mi*16 + r + 8*hi)*HID_
            + n0 + wn*32 + ni*16 + nc] = acc[mi][ni][r];
}

// ============================================================================
extern "C" void kernel_launch(void* const* d_in, const int* in_sizes, int n_in,
                              void* d_out, int out_size, void* d_ws, size_t ws_size,
                              hipStream_t stream) {
  const float*         hidden = (const float*)d_in[0];
  const unsigned char* mask   = (const unsigned char*)d_in[1];
  const int*           pos    = (const int*)d_in[2];
  const float*         Wqkv   = (const float*)d_in[3];
  const float*         Wo     = (const float*)d_in[4];
  float*               out    = (float*)d_out;

  const size_t headElems = (size_t)B_ * H_ * S_ * D_;
  __bf16* Qr  = (__bf16*)d_ws;
  __bf16* Kr  = Qr + headElems;
  __bf16* Vt  = Kr + headElems;
  __bf16* ctx = Vt + headElems;

  qkv_rope_kernel<<<dim3(96, 32), 256, 0, stream>>>(hidden, pos, Wqkv, Qr, Kr, Vt);
  attention_kernel<<<dim3(S_/64, B_*H_), 256, 0, stream>>>(Qr, Kr, Vt, mask, ctx);
  out_proj_kernel<<<dim3(HID_/64, (B_*S_)/128), 256, 0, stream>>>(ctx, Wo, out);
}
